// ImagePairCrossAttention_54314156425561
// MI455X (gfx1250) — matrix operand order repaired and checked by reading, not compiled, not run on hardware
//
#include <hip/hip_runtime.h>

typedef __attribute__((ext_vector_type(16))) _Float16 v16h;
typedef __attribute__((ext_vector_type(8)))  _Float16 v8h;
typedef __attribute__((ext_vector_type(8)))  float    v8f;
typedef __attribute__((ext_vector_type(4)))  int      v4i;

#define C_DIM 256
#define N_TOK 1024   // 32*32 spatial
#define B_DIM 32
#define NEG_BIG (-1e30f)
#define KPAD 264     // padded LDS row stride (halves)

#if __has_builtin(__builtin_amdgcn_global_load_async_to_lds_b128)
#define ATTN_ASYNC 1
#else
#define ATTN_ASYNC 0
#endif

union AHalves { v16h v; v8h h[2]; };

static __device__ __forceinline__ v8f wmma_f16(v16h a, v16h b, v8f c) {
    return __builtin_amdgcn_wmma_f32_16x16x32_f16(false, a, false, b, (short)0, c, false, false);
}

// ---------------------------------------------------------------------------
// Kernel 0: convert Wq / Wk (f32 row-major [C,C]) to f16 in workspace
// ---------------------------------------------------------------------------
__global__ void convert_weights_kernel(const float* __restrict__ Wq,
                                       const float* __restrict__ Wk,
                                       _Float16* __restrict__ WqH,
                                       _Float16* __restrict__ WkH) {
    int t = blockIdx.x * blockDim.x + threadIdx.x;   // 65536 threads
    WqH[t] = (_Float16)Wq[t];
    WkH[t] = (_Float16)Wk[t];
}

// ---------------------------------------------------------------------------
// Kernel 1: fused LayerNorm + projection (WMMA).
//   in  : x [B, C, 1024] f32  (token n lives at x[(b*C + c)*1024 + n])
//   out : Qh [B*1024, 256] f16, rows = tokens, scaled by `scale`
// Block = 256 threads (8 waves), handles (b, 32 tokens).
// ---------------------------------------------------------------------------
__global__ void ln_proj_kernel(const float* __restrict__ x,
                               const float* __restrict__ gamma,
                               const float* __restrict__ beta,
                               const _Float16* __restrict__ Wh,   // [256,256] f16
                               const float* __restrict__ bias,    // [256]
                               _Float16* __restrict__ outQ,       // [32768,256] f16
                               float scale) {
    __shared__ float    sf32[32 * 263];          // [token][c], pad 263 (bank-safe)
    __shared__ _Float16 s16 [32 * 264];          // LN'd f16   [token][c], pad 264

    const int b    = blockIdx.x >> 5;            // 0..31
    const int n0   = (blockIdx.x & 31) << 5;     // 0,32,...,992
    const int wv   = threadIdx.x >> 5;           // wave 0..7
    const int lane = threadIdx.x & 31;

    // ---- stage: coalesced load of [256c x 32n] tile, transpose into LDS ----
    {
        const int cbase = wv * 32;
#pragma unroll
        for (int i = 0; i < 32; ++i) {
            int c = cbase + i;
            float v = x[((b * C_DIM + c) << 10) + n0 + lane];
            sf32[lane * 263 + c] = v;            // row = token-in-tile (=lane)
        }
    }
    __syncthreads();

    // ---- LayerNorm: each wave normalizes 4 tokens (wave-wide reduction) ----
#pragma unroll
    for (int tt = 0; tt < 4; ++tt) {
        int tok  = wv * 4 + tt;
        int base = tok * 263 + lane * 8;
        float vals[8];
        float s = 0.f, ss = 0.f;
#pragma unroll
        for (int j = 0; j < 8; ++j) {
            float v = sf32[base + j];
            vals[j] = v; s += v; ss += v * v;
        }
#pragma unroll
        for (int off = 16; off >= 1; off >>= 1) {
            s  += __shfl_xor(s,  off, 32);
            ss += __shfl_xor(ss, off, 32);
        }
        float mu = s * (1.f / C_DIM);
        float var = ss * (1.f / C_DIM) - mu * mu;
        float rs = rsqrtf(var + 1e-5f);
#pragma unroll
        for (int j = 0; j < 8; ++j) {
            int c = lane * 8 + j;
            float h = (vals[j] - mu) * rs * gamma[c] + beta[c];
            s16[tok * 264 + c] = (_Float16)h;
        }
    }
    __syncthreads();

    // ---- WMMA GEMM: wave -> 16 tokens x 64 out-channels ----
    const int t    = wv & 1;                     // token sub-tile (0/1)
    const int ocb0 = (wv >> 1) << 6;             // out-channel base (0,64,128,192)
    const int lm   = lane & 15;
    const int hi   = lane >> 4;                  // 0 / 1

    v8f acc[4];
#pragma unroll
    for (int s = 0; s < 4; ++s) {
        float bv = bias[ocb0 + s * 16 + lm];
#pragma unroll
        for (int e = 0; e < 8; ++e) acc[s][e] = bv;
    }

#pragma unroll
    for (int k = 0; k < 8; ++k) {
        // A: LN'd tokens from LDS. lane<16: K {0..7,16..23}; lane>=16: {8..15,24..31}
        AHalves a;
        int hbase = (t * 16 + lm) * 264 + k * 32 + hi * 8;
        a.h[0] = *reinterpret_cast<const v8h*>(&s16[hbase]);
        a.h[1] = *reinterpret_cast<const v8h*>(&s16[hbase + 16]);
#pragma unroll
        for (int s = 0; s < 4; ++s) {
            // B[kk][n] = W[oc_n][c_kk]: lane reads K {8hi..8hi+7, 16+8hi..16+8hi+7} of row oc
            int brow = ocb0 + s * 16 + lm;
            int bcol = k * 32 + hi * 8;
            AHalves bu;
            bu.h[0] = *reinterpret_cast<const v8h*>(&Wh[brow * C_DIM + bcol]);
            bu.h[1] = *reinterpret_cast<const v8h*>(&Wh[brow * C_DIM + bcol + 16]);
            v16h bm = bu.v;
            acc[s] = wmma_f16(a.v, bm, acc[s]);
        }
    }

    // ---- store f16 (scaled): lane holds col oc, rows m = v + 8*hi ----
    const int tokbase = b * N_TOK + n0 + t * 16;
#pragma unroll
    for (int s = 0; s < 4; ++s) {
        int oc = ocb0 + s * 16 + lm;
#pragma unroll
        for (int v = 0; v < 8; ++v) {
            int m = v + (hi << 3);
            outQ[(size_t)(tokbase + m) * C_DIM + oc] = (_Float16)(acc[s][v] * scale);
        }
    }
}

// ---------------------------------------------------------------------------
// K-tile staging: 16 rows x 512B global -> LDS rows padded to 528B.
// 256 threads x 2 x 16B = 8KB. Async-to-LDS when available (ASYNCcnt path).
// ---------------------------------------------------------------------------
static __device__ __forceinline__ void stage_tile(const _Float16* __restrict__ g,
                                                  _Float16* __restrict__ l, int tid) {
#pragma unroll
    for (int s = 0; s < 2; ++s) {
        int f   = tid * 2 + s;                   // 0..511 chunks of 16B
        int row = f >> 5;
        int col = (f & 31) << 3;                 // halves
        const _Float16* gp = g + row * C_DIM + col;
        _Float16*       lp = l + row * KPAD + col;
#if ATTN_ASYNC
        __builtin_amdgcn_global_load_async_to_lds_b128(
            (__attribute__((address_space(1))) v4i*)gp,
            (__attribute__((address_space(3))) v4i*)lp, 0, 0);
#else
        *reinterpret_cast<v8h*>(lp) = *reinterpret_cast<const v8h*>(gp);
#endif
    }
}

static __device__ __forceinline__ void async_stage_wait() {
#if ATTN_ASYNC
#if __has_builtin(__builtin_amdgcn_s_wait_asynccnt)
    __builtin_amdgcn_s_wait_asynccnt(0);
#else
    asm volatile("s_wait_asynccnt 0" ::: "memory");
#endif
#endif
}

static __device__ __forceinline__ void online_update(const v8f& acc, float& m, float& l,
                                                     float& dg, bool isdiag, int didx) {
    float t0 = acc[0];
#pragma unroll
    for (int v = 1; v < 8; ++v) t0 = fmaxf(t0, acc[v]);
    float tmax = fmaxf(t0, __shfl_xor(t0, 16, 32));
    float nm = fmaxf(m, tmax);
    float p = 0.f;
#pragma unroll
    for (int v = 0; v < 8; ++v) p += __expf(acc[v] - nm);
    l = l * __expf(m - nm) + p;
    m = nm;
    if (isdiag && didx >= 0 && didx < 8) {
#pragma unroll
        for (int v = 0; v < 8; ++v) dg = (didx == v) ? acc[v] : dg;
    }
}

// ---------------------------------------------------------------------------
// Kernel 2: online-softmax diagonal.
// Block = 256 threads (8 waves), one batch b, 256 queries (wave = 2 q-tiles).
// K tiles double-buffered in LDS (async-to-LDS), shared by all 8 waves.
// S^T tiles: A = K-tile rows (keys), B = Q-tile cols (queries); softmax
// over keys = per-lane over 8 acc VGPRs + shfl_xor(16) with partner lane.
// ---------------------------------------------------------------------------
__global__ void attn_diag_kernel(const _Float16* __restrict__ Qh,
                                 const _Float16* __restrict__ Kh,
                                 float* __restrict__ out) {
    __shared__ _Float16 kbuf[2][16 * KPAD];

    const int tid  = threadIdx.x;
    const int wv   = tid >> 5;
    const int lane = tid & 31;
    const int lm   = lane & 15;
    const int hi   = lane >> 4;

    const int b     = blockIdx.x >> 2;                     // 0..31
    const int qbase = ((blockIdx.x & 3) << 8) + (wv << 5); // wave: 32 queries
    const int q0 = qbase, q1 = qbase + 16;

    const _Float16* Kb = Kh + (size_t)b * N_TOK * C_DIM;

    // Preload Q B-fragments for both q-tiles (reused across all 64 key tiles)
    v16h bq0[8], bq1[8];
    {
        const _Float16* qr0 = Qh + (size_t)(b * N_TOK + q0 + lm) * C_DIM + hi * 8;
        const _Float16* qr1 = qr0 + 16 * C_DIM;
#pragma unroll
        for (int c = 0; c < 8; ++c) {
            AHalves u0, u1;
            u0.h[0] = *reinterpret_cast<const v8h*>(qr0 + c * 32);
            u0.h[1] = *reinterpret_cast<const v8h*>(qr0 + c * 32 + 16);
            u1.h[0] = *reinterpret_cast<const v8h*>(qr1 + c * 32);
            u1.h[1] = *reinterpret_cast<const v8h*>(qr1 + c * 32 + 16);
            bq0[c] = u0.v;
            bq1[c] = u1.v;
        }
    }

    float m0 = NEG_BIG, l0 = 0.f, d0 = NEG_BIG;
    float m1 = NEG_BIG, l1 = 0.f, d1 = NEG_BIG;
    const int dt0 = q0 >> 4, dt1 = q1 >> 4;
    const int didx = lm - (hi << 3);             // lane holds diag iff 0<=didx<8

    stage_tile(Kb, &kbuf[0][0], tid);

    for (int kt = 0; kt < 64; ++kt) {
        async_stage_wait();
        __syncthreads();                          // tile kt ready; prev reads done
        if (kt + 1 < 64)
            stage_tile(Kb + (size_t)(kt + 1) * 16 * C_DIM, &kbuf[(kt + 1) & 1][0], tid);

        const _Float16* lsrc = &kbuf[kt & 1][0] + lm * KPAD + hi * 8;
        v8f acc0 = {}, acc1 = {};
#pragma unroll
        for (int c = 0; c < 8; ++c) {
            AHalves a;
            a.h[0] = *reinterpret_cast<const v8h*>(lsrc + c * 32);
            a.h[1] = *reinterpret_cast<const v8h*>(lsrc + c * 32 + 16);
            acc0 = wmma_f16(a.v, bq0[c], acc0);
            acc1 = wmma_f16(a.v, bq1[c], acc1);
        }
        online_update(acc0, m0, l0, d0, kt == dt0, didx);
        online_update(acc1, m1, l1, d1, kt == dt1, didx);
    }

    float lt0 = l0 + __shfl_xor(l0, 16, 32);
    float lt1 = l1 + __shfl_xor(l1, 16, 32);
    float g0  = fmaxf(d0, __shfl_xor(d0, 16, 32));
    float g1  = fmaxf(d1, __shfl_xor(d1, 16, 32));
    if (lane < 16) {
        out[b * N_TOK + q0 + lane] = __expf(g0 - m0) / lt0;
        out[b * N_TOK + q1 + lane] = __expf(g1 - m1) / lt1;
    }
}

// ---------------------------------------------------------------------------
extern "C" void kernel_launch(void* const* d_in, const int* in_sizes, int n_in,
                              void* d_out, int out_size, void* d_ws, size_t ws_size,
                              hipStream_t stream) {
    const float* x     = (const float*)d_in[0];
    const float* y     = (const float*)d_in[1];
    const float* gamma = (const float*)d_in[2];
    const float* beta  = (const float*)d_in[3];
    const float* Wq    = (const float*)d_in[4];
    const float* Wk    = (const float*)d_in[5];
    const float* bq    = (const float*)d_in[6];
    const float* bk    = (const float*)d_in[7];
    float* out = (float*)d_out;

    char* ws = (char*)d_ws;
    const size_t qkBytes = (size_t)B_DIM * N_TOK * C_DIM * sizeof(_Float16); // 16 MB
    _Float16* Qh  = (_Float16*)(ws);
    _Float16* Kh  = (_Float16*)(ws + qkBytes);
    _Float16* WqH = (_Float16*)(ws + 2 * qkBytes);
    _Float16* WkH = (_Float16*)(ws + 2 * qkBytes + (size_t)C_DIM * C_DIM * sizeof(_Float16));

    convert_weights_kernel<<<C_DIM * C_DIM / 256, 256, 0, stream>>>(Wq, Wk, WqH, WkH);

    const float scale = 0.0625f;  // 1/sqrt(256), folded into Q
    ln_proj_kernel<<<B_DIM * 32, 256, 0, stream>>>(x, gamma, beta, WqH, bq, Qh, scale);
    ln_proj_kernel<<<B_DIM * 32, 256, 0, stream>>>(y, gamma, beta, WkH, bk, Kh, 1.0f);

    attn_diag_kernel<<<B_DIM * 4, 256, 0, stream>>>(Qh, Kh, out);
}
